// RNNDecoder_67259187855569
// MI455X (gfx1250) — matrix operand order repaired and checked by reading, not compiled, not run on hardware
//
#include <hip/hip_runtime.h>

typedef __bf16 bf16;
typedef __attribute__((ext_vector_type(16))) __bf16 v16bf;
typedef __attribute__((ext_vector_type(8)))  __bf16 bf16x8;
typedef __attribute__((ext_vector_type(8)))  float  v8f;

#define HDIM   128
#define TLEN   256
#define BT     16        // batch rows per workgroup
#define NBLK   64        // 1024 / 16
#define NTHR   256       // 8 waves -> 2 waves per SIMD32
#define FPAD   132       // padded f32 row (128+4) -> bank shift, 16B-aligned stride
#define BPAD   136       // padded bf16 row (128+8) -> 16B aligned, bank shift
#define FRAG   512       // 32 lanes * 16 bf16 per WMMA B fragment

// ---------- fragment loaders ----------
__device__ __forceinline__ v16bf load_frag16(const bf16* p0, const bf16* p1) {
  union { v16bf v; bf16x8 h[2]; } u;
  u.h[0] = *(const bf16x8*)p0;
  u.h[1] = *(const bf16x8*)p1;
  return u.v;
}
// A-matrix 16x32 bf16: lanes 0-15 -> K {0..15}; lanes 16-31 -> K {16..31}
__device__ __forceinline__ v16bf load_a_frag(const bf16* rowp, int kc, int lane) {
  const bf16* p = rowp + kc * 32 + ((lane & 16) ? 16 : 0);
  return load_frag16(p, p + 8);
}
// B-matrix 32x16 bf16, pre-packed: lane n holds its 16 contiguous K values
__device__ __forceinline__ v16bf load_b_frag(const bf16* frag, int lane) {
  const bf16* p = frag + lane * 16;
  return load_frag16(p, p + 8);
}

__device__ __forceinline__ float sigmoidf(float x) {
  return 1.f / (1.f + __expf(-x));
}
__device__ __forceinline__ float tanh_fast(float x) {
  x = fminf(fmaxf(x, -15.f), 15.f);
  float e = __expf(2.f * x);
  return (e - 1.f) / (e + 1.f);
}

// ---------- weight pre-pack: f32 [3][384][128] -> bf16 WMMA B fragments ----------
// frag id = ((l*2+m)*24 + ct)*4 + kc ; element = lane*16 + e
// lane<16: K = kc*32 + e ; lane>=16: K = kc*32 + 16 + e ; column j = ct*16 + (lane&15)
__global__ void gru_pack_weights(const float* __restrict__ Wih,
                                 const float* __restrict__ Whh,
                                 bf16* __restrict__ packed) {
  int idx = blockIdx.x * blockDim.x + threadIdx.x;
  const int total = 3 * 2 * 24 * 4 * FRAG;   // 294912
  if (idx >= total) return;
  int e  = idx & 15;
  int ln = (idx >> 4) & 31;
  int kc = (idx >> 9) & 3;
  int fr = idx >> 11;            // (l*2+m)*24 + ct
  int ct = fr % 24;
  int lm = fr / 24;
  int l  = lm >> 1, m = lm & 1;
  int j  = ct * 16 + (ln & 15);
  int K  = kc * 32 + ((ln & 16) ? 16 : 0) + e;
  const float* W = m ? Whh : Wih;
  packed[idx] = (bf16)W[(l * 384 + j) * HDIM + K];
}

// ---------- persistent GRU kernel: one WG = 16 batch rows, full sequence ----------
__global__ __launch_bounds__(NTHR) void gru_seq_kernel(
    const float* __restrict__ hiddens,   // [B,3,H]
    const float* __restrict__ b_ih,      // [3,384]
    const float* __restrict__ b_hh,      // [3,384]
    const float* __restrict__ fc_w,      // [2,128]
    const float* __restrict__ fc_b,      // [2]
    const bf16*  __restrict__ packW,     // packed fragments
    float* __restrict__ out)             // [B,T,2]
{
  extern __shared__ char smem[];
  float* r_lds = (float*)smem;                       // 16*132
  float* z_lds = r_lds + BT * FPAD;                  // 16*132
  float* hf32  = z_lds + BT * FPAD;                  // 3*16*132
  float* bihL  = hf32 + 3 * BT * FPAD;               // 3*384
  float* bhhL  = bihL + 3 * 384;                     // 3*384
  float* fcwL  = bhhL + 3 * 384;                     // 2*128 (16B aligned)
  float* fcbL  = fcwL + 2 * HDIM;                    // 4 (padded)
  bf16*  hbf   = (bf16*)(fcbL + 4);                  // 3*16*136
  bf16*  zbuf  = hbf + 3 * BT * BPAD;                // 16*136

  const int tid   = threadIdx.x;
  const int lane  = tid & 31;
  const int w     = tid >> 5;        // 0..7
  const int nlane = lane & 15;
  const int rbase = (lane & 16) ? 8 : 0;
  const int b0    = blockIdx.x * BT;

  // ---- init: biases, FC params, zero input buffer, hidden state ----
  for (int i = tid; i < 3 * 384; i += NTHR) { bihL[i] = b_ih[i]; bhhL[i] = b_hh[i]; }
  for (int i = tid; i < 2 * HDIM; i += NTHR) fcwL[i] = fc_w[i];
  if (tid < 2) fcbL[tid] = fc_b[tid];
  for (int i = tid; i < BT * BPAD; i += NTHR) zbuf[i] = (bf16)0.f;
  for (int i = tid; i < 3 * BT * HDIM; i += NTHR) {
    int k = i & 127, row = (i >> 7) & 15, l = i >> 11;
    float v = hiddens[(size_t)(b0 + row) * (3 * HDIM) + l * HDIM + k];
    hf32[(l * BT + row) * FPAD + k] = v;
    hbf [(l * BT + row) * BPAD + k] = (bf16)v;
  }
  __syncthreads();

  for (int t = 0; t < TLEN; ++t) {
    for (int l = 0; l < 3; ++l) {
      const bf16* xbf = (l == 0) ? ((t == 0) ? zbuf : (hbf + 2 * BT * BPAD))
                                 : (hbf + (l - 1) * BT * BPAD);
      bf16*  hbfl  = hbf  + l * BT * BPAD;
      float* hf32l = hf32 + l * BT * FPAD;
      const bf16* wih = packW + (size_t)((l * 2 + 0) * 96) * FRAG;
      const bf16* whh = packW + (size_t)((l * 2 + 1) * 96) * FRAG;

      const bf16* xrow = xbf  + nlane * BPAD;
      const bf16* hrow = hbfl + nlane * BPAD;

      v8f accx_n, acch_n;
      // column-tile schedule: wave w owns ct = w (r), w+8 (z), w+16 (n)
      #pragma unroll
      for (int i = 0; i < 3; ++i) {
        const int ct = w + 8 * i;
        const float bx = bihL[l * 384 + ct * 16 + nlane];
        const float bh = bhhL[l * 384 + ct * 16 + nlane];
        v8f ax = {bx, bx, bx, bx, bx, bx, bx, bx};
        v8f ah = {bh, bh, bh, bh, bh, bh, bh, bh};
        #pragma unroll
        for (int kc = 0; kc < 4; ++kc) {
          v16bf a_x  = load_a_frag(xrow, kc, lane);
          v16bf a_h  = load_a_frag(hrow, kc, lane);
          v16bf b_iw = load_b_frag(wih + (size_t)(ct * 4 + kc) * FRAG, lane);
          v16bf b_hw = load_b_frag(whh + (size_t)(ct * 4 + kc) * FRAG, lane);
          ax = __builtin_amdgcn_wmma_f32_16x16x32_bf16(false, a_x, false, b_iw,
                                                       (short)0, ax, false, false);
          ah = __builtin_amdgcn_wmma_f32_16x16x32_bf16(false, a_h, false, b_hw,
                                                       (short)0, ah, false, false);
        }
        if (i < 2) {
          float* gbuf = (i == 0) ? r_lds : z_lds;
          const int lcol = ct * 16 + nlane - ((i == 0) ? 0 : 128);
          #pragma unroll
          for (int v = 0; v < 8; ++v)
            gbuf[(v + rbase) * FPAD + lcol] = sigmoidf(ax[v] + ah[v]);
        } else {
          accx_n = ax;
          acch_n = ah;
        }
      }
      __syncthreads();   // r,z visible; all reads of x/h bf16 complete

      {
        const int jj = w * 16 + nlane;            // hidden index 0..127
        #pragma unroll
        for (int v = 0; v < 8; ++v) {
          const int row = v + rbase;
          float r    = r_lds[row * FPAD + jj];
          float z    = z_lds[row * FPAD + jj];
          float nval = tanh_fast(accx_n[v] + r * acch_n[v]);
          float hold = hf32l[row * FPAD + jj];
          float hnew = (1.f - z) * nval + z * hold;
          hf32l[row * FPAD + jj] = hnew;
          hbfl [row * BPAD + jj] = (bf16)hnew;
        }
      }
      __syncthreads();   // new h visible to next layer / next step
    }

    // ---- inline FC from LDS: out[b,t,o] = fc_w[o,:] . h_top[b,:] + fc_b[o] ----
    if (tid < 32) {
      const int o   = tid & 1;
      const int row = tid >> 1;
      const float4* hv = (const float4*)(hf32 + 2 * BT * FPAD + row * FPAD);
      const float4* wv = (const float4*)(fcwL + o * HDIM);
      float s = fcbL[o];
      #pragma unroll
      for (int k = 0; k < HDIM / 4; ++k) {
        float4 a = wv[k], h = hv[k];
        s += a.x * h.x + a.y * h.y + a.z * h.z + a.w * h.w;
      }
      out[((size_t)(b0 + row) * TLEN + t) * 2 + o] = s;
    }
    __syncthreads();
  }
}

extern "C" void kernel_launch(void* const* d_in, const int* in_sizes, int n_in,
                              void* d_out, int out_size, void* d_ws, size_t ws_size,
                              hipStream_t stream) {
  const float* hiddens = (const float*)d_in[0];
  const float* W_ih    = (const float*)d_in[1];
  const float* W_hh    = (const float*)d_in[2];
  const float* b_ih    = (const float*)d_in[3];
  const float* b_hh    = (const float*)d_in[4];
  const float* fc_w    = (const float*)d_in[5];
  const float* fc_b    = (const float*)d_in[6];
  bf16* packW = (bf16*)d_ws;   // needs 3*2*24*4*512*2 = 589824 bytes

  const int packN = 3 * 2 * 24 * 4 * FRAG;
  gru_pack_weights<<<(packN + 255) / 256, 256, 0, stream>>>(W_ih, W_hh, packW);

  const size_t lds_bytes =
      (size_t)(2 * BT * FPAD + 3 * BT * FPAD + 2 * 3 * 384 + 2 * HDIM + 4) * sizeof(float) +
      (size_t)(3 * BT * BPAD + BT * BPAD) * sizeof(bf16);   // 69904 B

  gru_seq_kernel<<<NBLK, NTHR, lds_bytes, stream>>>(
      hiddens, b_ih, b_hh, fc_w, fc_b, packW, (float*)d_out);
}
